// MultiHeadAttention_7035156430929
// MI455X (gfx1250) — matrix operand order repaired and checked by reading, not compiled, not run on hardware
//
#include <hip/hip_runtime.h>
#include <hip/hip_bf16.h>

// Problem constants (from reference): B=4, S=1024, E=1024, H=16, D=64
#define PB 4
#define PS 1024
#define PE 1024
#define PH 16
#define PD 64

// LDS weight-panel row pitch in bf16 elements (64B data + 16B pad = 20-bank step,
// so the 16 lanes of a B-fragment ds_load_b128 hit 16 distinct banks).
#define WPITCH 40

typedef __attribute__((ext_vector_type(16))) __bf16 v16bf;
typedef __attribute__((ext_vector_type(8)))  float  v8f;

union Frag {
    v16bf v;
    unsigned short u[16];
    uint4 q[2];
};

__device__ __forceinline__ unsigned short f2bf(float x) {
    union { float f; unsigned u; } cv; cv.f = x;
    unsigned u = cv.u + (0x7FFFu + ((cv.u >> 16) & 1u));   // round-to-nearest-even
    return (unsigned short)(u >> 16);
}

// ---------------------------------------------------------------------------
// A fragment: 16x32 bf16 (M x K), contiguous k order (matches B loaders):
//   lane<16 : row M=lane,    elems 0..15 = K 0..15
//   lane>=16: row M=lane-16, elems 0..15 = K 16..31
// ---------------------------------------------------------------------------
__device__ __forceinline__ v16bf load_a_bf16(const unsigned short* __restrict__ p,
                                             int ld, int m0, int kb, int lane) {
    int half = lane >> 4;
    int r    = lane & 15;
    const unsigned short* s0 = p + (size_t)(m0 + r) * ld + kb + half * 16;
    Frag f;
    f.q[0] = *(const uint4*)(s0);
    f.q[1] = *(const uint4*)(s0 + 8);
    return f.v;
}

// ---------------------------------------------------------------------------
// B fragment from global: 32x16 (K x N), element (k,n) = W[n][k], k contiguous.
//   lane<16 : col N=lane,    elems 0..15 = K 0..15
//   lane>=16: col N=lane-16, elems 0..15 = K 16..31
// ---------------------------------------------------------------------------
__device__ __forceinline__ v16bf load_b_bf16(const unsigned short* __restrict__ p,
                                             int ld, int n0, int kb, int lane) {
    int kh = lane >> 4;
    int n  = lane & 15;
    const unsigned short* s = p + (size_t)(n0 + n) * ld + kb + kh * 16;
    Frag f;
    f.q[0] = ((const uint4*)s)[0];
    f.q[1] = ((const uint4*)s)[1];
    return f.v;
}

// B fragment from an LDS weight panel laid out [64 rows][WPITCH halves].
__device__ __forceinline__ v16bf load_b_lds(const unsigned short* sm, int nt, int lane) {
    int kh = lane >> 4;
    int n  = lane & 15;
    const unsigned short* s = sm + (nt * 16 + n) * WPITCH + kh * 16;
    Frag f;
    f.q[0] = ((const uint4*)s)[0];
    f.q[1] = ((const uint4*)s)[1];
    return f.v;
}

__device__ __forceinline__ v8f wmma_bf16(v16bf a, v16bf b, v8f c) {
    return __builtin_amdgcn_wmma_f32_16x16x32_bf16(false, a, false, b, (short)0, c,
                                                   false, false);
}

// ---------------------------------------------------------------------------
// Async global->LDS staging of a 64(N) x 32(K) bf16 weight panel (4 KB data,
// padded rows). 256 16-byte transfers, 128 threads -> 2 async b128 per thread
// (2 ASYNCcnt increments per wave per panel).
// ---------------------------------------------------------------------------
__device__ __forceinline__ void issue_async_panel(const unsigned short* __restrict__ Wsrc,
                                                  int n0, int kb, unsigned lds_base,
                                                  int tid) {
#pragma unroll
    for (int i = 0; i < 2; ++i) {
        int t     = tid + i * 128;          // 0..255
        int row   = t >> 2;                 // 0..63
        int chunk = t & 3;                  // 4 x 16B per 64B row
        unsigned lds_off = lds_base + (unsigned)(row * (WPITCH * 2) + chunk * 16);
        unsigned long long ga =
            (unsigned long long)(const void*)(Wsrc + (size_t)(n0 + row) * PE + kb + chunk * 8);
        asm volatile("global_load_async_to_lds_b128 %0, %1, off"
                     :: "v"(lds_off), "v"(ga) : "memory");
    }
}

// ---------------------------------------------------------------------------
// f32 -> bf16 conversion pass
// ---------------------------------------------------------------------------
__global__ void cvt_f32_bf16(const float* __restrict__ src,
                             unsigned short* __restrict__ dst, int n) {
    int i = blockIdx.x * blockDim.x + threadIdx.x;
    if (i < n) dst[i] = f2bf(src[i]);
}

// ---------------------------------------------------------------------------
// Projection: y[m,n] = sum_k X[m,k]*W[n,k] + bias[n].
// 128-thread (4-wave) block computes a 64x64 tile; the W panel for each k-step
// is async-staged into a double-buffered LDS panel shared by all 4 waves.
//   transposeV==0 : out bf16 [B,H,S,D]   (Q, K)
//   transposeV==1 : out bf16 [B,H,D,S]   (V, pre-transposed for the PV GEMM)
// ---------------------------------------------------------------------------
__global__ void __launch_bounds__(128)
proj_kernel(const unsigned short* __restrict__ X,
            const unsigned short* __restrict__ W,
            const float* __restrict__ bias,
            unsigned short* __restrict__ out,
            int transposeV) {
    __shared__ unsigned short wt[2][64 * WPITCH];   // 2 x 5120 B

    const int tid  = threadIdx.x;
    const int lane = tid & 31;
    const int wv   = tid >> 5;
    const int m0 = blockIdx.x * 64 + wv * 16;
    const int n0 = blockIdx.y * 64;

    const unsigned lb0 = (unsigned)(uintptr_t)&wt[0][0];
    const unsigned lb1 = (unsigned)(uintptr_t)&wt[1][0];

    issue_async_panel(W, n0, 0, lb0, tid);

    v8f acc[4] = {};
    for (int kb = 0, it = 0; kb < PE; kb += 32, ++it) {
        int cur = it & 1;
        if (kb + 32 < PE) {
            issue_async_panel(W, n0, kb + 32, cur ? lb0 : lb1, tid);
            asm volatile("s_wait_asynccnt 0x2" ::: "memory");   // current panel landed
        } else {
            asm volatile("s_wait_asynccnt 0x0" ::: "memory");
        }
        __syncthreads();                                        // all waves' panels visible
        v16bf a = load_a_bf16(X, PE, m0, kb, lane);
#pragma unroll
        for (int t = 0; t < 4; ++t) {
            v16bf b = load_b_lds(&wt[cur][0], t, lane);
            acc[t] = wmma_bf16(a, b, acc[t]);
        }
        __syncthreads();                                        // reads done before rewrite
    }

    const int col   = lane & 15;
    const int rbase = (lane >> 4) * 8;
#pragma unroll
    for (int t = 0; t < 4; ++t) {
        int n = n0 + 16 * t + col;
        float bv = bias[n];
        int hh = n >> 6;   // head
        int dd = n & 63;   // dim within head
#pragma unroll
        for (int r = 0; r < 8; ++r) {
            int m = m0 + rbase + r;
            int bb = m >> 10;     // batch (S == 1024)
            int ss = m & 1023;    // seq pos
            float y = acc[t][r] + bv;
            size_t idx = transposeV
                ? (((size_t)(bb * PH + hh) * PD + dd) * PS + ss)   // [B,H,D,S]
                : (((size_t)(bb * PH + hh) * PS + ss) * PD + dd);  // [B,H,S,D]
            out[idx] = f2bf(y);
        }
    }
}

// ---------------------------------------------------------------------------
// Attention core: one wave per (b, h, 16-row q tile).
//   scores = (Q Kt) * 1/sqrt(E), masked, softmax over 1024 keys (LDS panel),
//   O = P V  (V pre-transposed to [B,H,D,S]).  Output bf16 [B,S,E].
// ---------------------------------------------------------------------------
__global__ void __launch_bounds__(32)
attn_kernel(const unsigned short* __restrict__ Qg,   // [B,H,S,D] bf16
            const unsigned short* __restrict__ Kg,   // [B,H,S,D] bf16
            const unsigned short* __restrict__ Vt,   // [B,H,D,S] bf16
            const int* __restrict__ mask,            // [B,S]
            unsigned short* __restrict__ O) {        // [B,S,E] bf16
    __shared__ float sc[16 * PS];                    // 64 KB score panel

    const int lane = threadIdx.x;
    const int qt = blockIdx.x;     // q tile, 0..63
    const int h  = blockIdx.y;
    const int b  = blockIdx.z;

    const unsigned short* Qbase = Qg + (size_t)(b * PH + h) * PS * PD;
    const unsigned short* Kbase = Kg + (size_t)(b * PH + h) * PS * PD;
    const unsigned short* Vbase = Vt + (size_t)(b * PH + h) * PD * PS;

    v16bf q0 = load_a_bf16(Qbase, PD, qt * 16, 0, lane);
    v16bf q1 = load_a_bf16(Qbase, PD, qt * 16, 32, lane);

    const float scale = 0.03125f;  // 1/sqrt(E) = 1/32
    const int col   = lane & 15;
    const int rbase = (lane >> 4) * 8;

    for (int j = 0; j < PS / 16; ++j) {
        v8f a = {};
        v16bf k0 = load_b_bf16(Kbase, PD, j * 16, 0, lane);
        v16bf k1 = load_b_bf16(Kbase, PD, j * 16, 32, lane);
        a = wmma_bf16(q0, k0, a);
        a = wmma_bf16(q1, k1, a);
        int kc = j * 16 + col;
        int mval = mask[b * PS + kc];
#pragma unroll
        for (int r = 0; r < 8; ++r) {
            float v = a[r] * scale;
            if (mval == 0) v = -1e20f;
            sc[(rbase + r) * PS + kc] = v;
        }
    }
    __syncthreads();

    {   // softmax: lane L handles row (L&15), half (L>>4); combine via shfl
        int row = lane & 15;
        int h0  = (lane >> 4) * 512;
        float* rp = &sc[row * PS + h0];
        float mx = -3.0e38f;
        for (int c = 0; c < 512; ++c) mx = fmaxf(mx, rp[c]);
        mx = fmaxf(mx, __shfl_xor(mx, 16, 32));
        float sum = 0.0f;
        for (int c = 0; c < 512; ++c) {
            float e = __expf(rp[c] - mx);
            rp[c] = e;
            sum += e;
        }
        sum += __shfl_xor(sum, 16, 32);
        float inv = 1.0f / sum;
        for (int c = 0; c < 512; ++c) rp[c] *= inv;
    }
    __syncthreads();

    // O = P @ V : K-loop over S=1024, N = D = 64 (4 n-tiles)
    v8f acco[4] = {};
    for (int kb = 0; kb < PS; kb += 32) {
        Frag af;
        {
            int half = lane >> 4;
            int r = lane & 15;
            const float* s0 = &sc[r * PS + kb + half * 16];
#pragma unroll
            for (int t = 0; t < 8; ++t) af.u[t] = f2bf(s0[t]);
#pragma unroll
            for (int t = 0; t < 8; ++t) af.u[8 + t] = f2bf(s0[8 + t]);
        }
#pragma unroll
        for (int n = 0; n < 4; ++n) {
            v16bf bv = load_b_bf16(Vbase, PS, n * 16, kb, lane);
            acco[n] = wmma_bf16(af.v, bv, acco[n]);
        }
    }

#pragma unroll
    for (int n = 0; n < 4; ++n) {
#pragma unroll
        for (int r = 0; r < 8; ++r) {
            int s = qt * 16 + rbase + r;
            int d = n * 16 + col;
            O[(size_t)(b * PS + s) * PE + h * PD + d] = f2bf(acco[n][r]);
        }
    }
}

// ---------------------------------------------------------------------------
// Output projection: out[m,n] = sum_k O[m,k]*Wo[n,k] + bo[n], f32 result.
// Same 4-wave async-LDS-staged structure as proj_kernel.
// ---------------------------------------------------------------------------
__global__ void __launch_bounds__(128)
outproj_kernel(const unsigned short* __restrict__ X,
               const unsigned short* __restrict__ W,
               const float* __restrict__ bias,
               float* __restrict__ out) {
    __shared__ unsigned short wt[2][64 * WPITCH];

    const int tid  = threadIdx.x;
    const int lane = tid & 31;
    const int wv   = tid >> 5;
    const int m0 = blockIdx.x * 64 + wv * 16;
    const int n0 = blockIdx.y * 64;

    const unsigned lb0 = (unsigned)(uintptr_t)&wt[0][0];
    const unsigned lb1 = (unsigned)(uintptr_t)&wt[1][0];

    issue_async_panel(W, n0, 0, lb0, tid);

    v8f acc[4] = {};
    for (int kb = 0, it = 0; kb < PE; kb += 32, ++it) {
        int cur = it & 1;
        if (kb + 32 < PE) {
            issue_async_panel(W, n0, kb + 32, cur ? lb0 : lb1, tid);
            asm volatile("s_wait_asynccnt 0x2" ::: "memory");
        } else {
            asm volatile("s_wait_asynccnt 0x0" ::: "memory");
        }
        __syncthreads();
        v16bf a = load_a_bf16(X, PE, m0, kb, lane);
#pragma unroll
        for (int t = 0; t < 4; ++t) {
            v16bf b = load_b_lds(&wt[cur][0], t, lane);
            acc[t] = wmma_bf16(a, b, acc[t]);
        }
        __syncthreads();
    }

    const int col   = lane & 15;
    const int rbase = (lane >> 4) * 8;
#pragma unroll
    for (int t = 0; t < 4; ++t) {
        int n = n0 + 16 * t + col;
        float bv = bias[n];
#pragma unroll
        for (int r = 0; r < 8; ++r) {
            int m = m0 + rbase + r;
            out[(size_t)m * PE + n] = acc[t][r] + bv;
        }
    }
}

// ---------------------------------------------------------------------------
extern "C" void kernel_launch(void* const* d_in, const int* in_sizes, int n_in,
                              void* d_out, int out_size, void* d_ws, size_t ws_size,
                              hipStream_t stream) {
    (void)in_sizes; (void)n_in; (void)out_size; (void)ws_size;

    const float* value  = (const float*)d_in[0];
    const float* key_in = (const float*)d_in[1];
    const float* query  = (const float*)d_in[2];
    const int*   mask   = (const int*)d_in[3];
    const float* Wq = (const float*)d_in[4];
    const float* bq = (const float*)d_in[5];
    const float* Wk = (const float*)d_in[6];
    const float* bk = (const float*)d_in[7];
    const float* Wv = (const float*)d_in[8];
    const float* bv = (const float*)d_in[9];
    const float* Wo = (const float*)d_in[10];
    const float* bo = (const float*)d_in[11];

    const size_t NBSE = (size_t)PB * PS * PE;   // 4,194,304
    const size_t NEE  = (size_t)PE * PE;        // 1,048,576

    unsigned short* wsu = (unsigned short*)d_ws;
    unsigned short* xq  = wsu;                  // bf16 query   [B*S,E]
    unsigned short* xk  = xq  + NBSE;           // bf16 key     [B*S,E]
    unsigned short* xv  = xk  + NBSE;           // bf16 value   [B*S,E]
    unsigned short* wqb = xv  + NBSE;           // bf16 Wq [E,E]
    unsigned short* wkb = wqb + NEE;
    unsigned short* wvb = wkb + NEE;
    unsigned short* wob = wvb + NEE;
    unsigned short* Qh  = wob + NEE;            // bf16 [B,H,S,D]
    unsigned short* Kh  = Qh  + NBSE;           // bf16 [B,H,S,D]
    unsigned short* Vth = Kh  + NBSE;           // bf16 [B,H,D,S]
    unsigned short* Oh  = xq;                   // reuse xq (dead after Q proj)

    // 1) f32 -> bf16 staging
    int nb = (int)((NBSE + 255) / 256);
    int nw = (int)((NEE + 255) / 256);
    cvt_f32_bf16<<<nb, 256, 0, stream>>>(query,  xq,  (int)NBSE);
    cvt_f32_bf16<<<nb, 256, 0, stream>>>(key_in, xk,  (int)NBSE);
    cvt_f32_bf16<<<nb, 256, 0, stream>>>(value,  xv,  (int)NBSE);
    cvt_f32_bf16<<<nw, 256, 0, stream>>>(Wq, wqb, (int)NEE);
    cvt_f32_bf16<<<nw, 256, 0, stream>>>(Wk, wkb, (int)NEE);
    cvt_f32_bf16<<<nw, 256, 0, stream>>>(Wv, wvb, (int)NEE);
    cvt_f32_bf16<<<nw, 256, 0, stream>>>(Wo, wob, (int)NEE);

    // 2) QKV projections (WMMA + async LDS weight staging). V transposed.
    dim3 pg((PB * PS) / 64, PE / 64);
    proj_kernel<<<pg, 128, 0, stream>>>(xq, wqb, bq, Qh, 0);
    proj_kernel<<<pg, 128, 0, stream>>>(xk, wkb, bk, Kh, 0);
    proj_kernel<<<pg, 128, 0, stream>>>(xv, wvb, bv, Vth, 1);

    // 3) attention: one wave per (b, h, 16-row q tile)
    attn_kernel<<<dim3(PS / 16, PH, PB), 32, 0, stream>>>(Qh, Kh, Vth, mask, Oh);

    // 4) output projection (f32 result)
    outproj_kernel<<<pg, 128, 0, stream>>>(Oh, wob, bo, (float*)d_out);
}
